// GAT_38405597561524
// MI455X (gfx1250) — compile-verified, matrix operand order checked by reading, not run on hardware
//
#include <hip/hip_runtime.h>
#include <cmath>

// ---------------------------------------------------------------------------
// GATv2 network for MI455X (gfx1250, wave32).
// Dense transforms: bf16 WMMA (v_wmma_f32_16x16x32_bf16), f32 accumulate,
// ping-pong software pipeline (B/A staged one k-chunk ahead in registers).
// Edge phases: wave-per-edge coalesced gather/scatter with L2 float atomics.
// ---------------------------------------------------------------------------

typedef __attribute__((ext_vector_type(16))) __bf16 v16bf;
typedef __attribute__((ext_vector_type(8)))  float  v8f;
typedef __attribute__((ext_vector_type(4)))  float  f32x4;

constexpr int  NN    = 50000;   // nodes
constexpr int  HC    = 192;     // HEADS*HID
constexpr int  XLD   = 768;     // xl|xr|xl_p|xr_p row stride
constexpr float NEG_SLOPE = 0.2f;

// ============================ dense GEMM (WMMA) ============================
// Out tile grid: blockIdx.x = 16-row stripe, blockIdx.y = column group of
// 8*CTW 16-col tiles; 8 waves per block, CTW tiles per wave.  KC is constexpr
// and the k-loop is fully unrolled, so all loads are constant-immediate
// offsets from two fixed base pointers.  b0/b1 (and q0/q1 for A) form a
// ping-pong pipeline: the load set for chunk kc+1 is issued before the WMMA
// burst of chunk kc, forcing overlapping live ranges (distinct registers)
// and a full burst of latency hiding per load set.
template <int CTW, int KC, int LDOUT>
__global__ __launch_bounds__(256)
void gemm_bf16_wmma(const float* __restrict__ A, const __bf16* __restrict__ Bp,
                    const float* __restrict__ bias, float* __restrict__ Out)
{
    constexpr int K = KC * 32;
    const int lane = threadIdx.x & 31;
    const int wave = threadIdx.x >> 5;
    const int l16  = lane & 15;
    const int half = lane >> 4;              // lanes 0-15 vs 16-31
    const long rowBase = (long)blockIdx.x * 16;
    const int ctBase = (blockIdx.y * 8 + wave) * CTW;

    // A fragment source: row (rowBase+l16); lanes 0-15 take K = {0..7,16..23},
    // lanes 16-31 take K = {8..15,24..31} of each 32-wide chunk (ISA 7.12.2).
    const char* aPtr = (const char*)(A + (rowBase + l16) * (long)K) + half * 32;
    // B fragments: per-lane 32B chunk; (t,kc) selected by constant offsets.
    const char* bPtr = (const char*)Bp + (size_t)ctBase * KC * 1024 + lane * 32;

    v8f c[CTW];
    const v8f zero = {0.f,0.f,0.f,0.f,0.f,0.f,0.f,0.f};
#pragma unroll
    for (int t = 0; t < CTW; ++t) c[t] = zero;

    v16bf b0[CTW], b1[CTW];
    f32x4 q0[4], q1[4];

    auto loadB = [&](v16bf (&b)[CTW], size_t byteOff) {
#pragma unroll
        for (int t = 0; t < CTW; ++t)
            b[t] = *(const v16bf*)(bPtr + byteOff + (size_t)t * (KC * 1024));
    };
    auto loadA = [&](f32x4 (&q)[4], size_t byteOff) {
        q[0] = *(const f32x4*)(aPtr + byteOff +  0);
        q[1] = *(const f32x4*)(aPtr + byteOff + 16);
        q[2] = *(const f32x4*)(aPtr + byteOff + 64);
        q[3] = *(const f32x4*)(aPtr + byteOff + 80);
    };
    auto cvtA = [&](const f32x4 (&q)[4]) {
        v16bf a;
#pragma unroll
        for (int i = 0; i < 4; ++i) {
            a[i]      = (__bf16)q[0][i];
            a[4 + i]  = (__bf16)q[1][i];
            a[8 + i]  = (__bf16)q[2][i];
            a[12 + i] = (__bf16)q[3][i];
        }
        return a;
    };
    auto burst = [&](const v16bf& a, v16bf (&b)[CTW]) {
#pragma unroll
        for (int t = 0; t < CTW; ++t)
            c[t] = __builtin_amdgcn_wmma_f32_16x16x32_bf16(
                       false, a, false, b[t], (short)0, c[t], false, false);
    };

    loadB(b0, 0);
    loadA(q0, 0);
#pragma unroll
    for (int kc = 0; kc < KC; kc += 2) {
        if (kc + 1 < KC) {                       // stage kc+1 while doing kc
            loadB(b1, (size_t)(kc + 1) * 1024);
            loadA(q1, (size_t)(kc + 1) * 128);
        }
        { const v16bf a = cvtA(q0); burst(a, b0); }
        if (kc + 1 < KC) {
            if (kc + 2 < KC) {                   // stage kc+2 while doing kc+1
                loadB(b0, (size_t)(kc + 2) * 1024);
                loadA(q0, (size_t)(kc + 2) * 128);
            }
            const v16bf a = cvtA(q1); burst(a, b1);
        }
    }
    // D layout: VGPR j -> row (half*8 + j), col = l16 within tile.
    float* outBase = Out + (rowBase + half * 8) * (long)LDOUT;
#pragma unroll
    for (int t = 0; t < CTW; ++t) {
        const int col = (ctBase + t) * 16 + l16;
        const float bi = bias[col];
#pragma unroll
        for (int j = 0; j < 8; ++j)
            outBase[(long)j * LDOUT + col] = c[t][j] + bi;
    }
}

// Pack W[K, ncolsSrc] (f32, row-major) into bf16 B-fragments at col-tile
// offset ctOffset of a GEMM with KCglob = K/32 k-chunks.
__global__ void pack_b(const float* __restrict__ W, __bf16* __restrict__ Bp,
                       int KCglob, int ncolsSrc, int ctOffset, long total)
{
    long idx = (long)blockIdx.x * 256 + threadIdx.x;
    if (idx >= total) return;
    const int i    = idx & 15;
    const int lane = (idx >> 4) & 31;
    const long rest = idx >> 9;
    const int kc  = (int)(rest % KCglob);
    const int ctl = (int)(rest / KCglob);
    const int half = lane >> 4;
    const int col  = ctl * 16 + (lane & 15);
    const int klocal = (i < 8) ? (half * 8 + i) : (16 + half * 8 + (i - 8));
    const int k = kc * 32 + klocal;
    Bp[(((long)(ctOffset + ctl) * KCglob + kc) * 32 + lane) * 16 + i] =
        (__bf16)W[(long)k * ncolsSrc + col];
}

__global__ void cat_bias4(float* __restrict__ dst,
                          const float* b0, const float* b1,
                          const float* b2, const float* b3)
{
    int i = blockIdx.x * 256 + threadIdx.x;
    if (i >= 4 * HC) return;
    const float* s = (i < HC) ? b0 : (i < 2 * HC) ? b1 : (i < 3 * HC) ? b2 : b3;
    dst[i] = s[i % HC];
}

__global__ void fill_f32(float* __restrict__ p, float v, long n)
{
    long i = (long)blockIdx.x * 256 + threadIdx.x;
    if (i < n) p[i] = v;
}

// ============================ edge phases ==================================
__device__ __forceinline__ void atomicMaxFloat(float* addr, float val)
{
    if (val >= 0.f) atomicMax((int*)addr, __float_as_int(val));
    else            atomicMin((unsigned int*)addr, __float_as_uint(val));
}

// Pass 1: per-edge GATv2 score  s[e,h] = att[h] . leaky(xl[src]+xr[dst]),
// plus segment max into m[dst,h].  One wave per edge; channel g = lane+32j
// so each step is a contiguous 128B gather across the wave.
__global__ __launch_bounds__(256)
void edge_score(const long long* __restrict__ ei, long E,
                const float* __restrict__ X, int xlOff, int xrOff,
                const float* __restrict__ att,
                float* __restrict__ score, float* __restrict__ m)
{
    const long e = (long)blockIdx.x * 8 + (threadIdx.x >> 5);
    if (e >= E) return;
    const int lane = threadIdx.x & 31;
    const long src = ei[e];
    const long dst = ei[E + e];
    const float* xl = X + src * (long)XLD + xlOff;
    const float* xr = X + dst * (long)XLD + xrOff;
    float acc0 = 0.f, acc1 = 0.f, acc2 = 0.f;
#pragma unroll
    for (int j = 0; j < 6; ++j) {                 // head h = j/2 (compile time)
        const int g = lane + 32 * j;
        const float s = xl[g] + xr[g];
        const float lr = s > 0.f ? s : NEG_SLOPE * s;
        const float contrib = lr * att[g];
        if (j < 2)      acc0 += contrib;
        else if (j < 4) acc1 += contrib;
        else            acc2 += contrib;
    }
    for (int off = 16; off; off >>= 1) {
        acc0 += __shfl_xor(acc0, off, 32);
        acc1 += __shfl_xor(acc1, off, 32);
        acc2 += __shfl_xor(acc2, off, 32);
    }
    if (lane == 0) {
        score[e * 3 + 0] = acc0;  atomicMaxFloat(&m[dst * 3 + 0], acc0);
        score[e * 3 + 1] = acc1;  atomicMaxFloat(&m[dst * 3 + 1], acc1);
        score[e * 3 + 2] = acc2;  atomicMaxFloat(&m[dst * 3 + 2], acc2);
    }
}

// Pass 2: ex = exp(score - m[dst]) (in place), segment-sum into denom.
__global__ void edge_exp(const long long* __restrict__ ei, long E,
                         const float* __restrict__ m,
                         float* __restrict__ score, float* __restrict__ denom)
{
    long idx = (long)blockIdx.x * 256 + threadIdx.x;
    if (idx >= E * 3) return;
    const long e = idx / 3;
    const int  h = (int)(idx - e * 3);
    const long dst = ei[E + e];
    const float ex = __expf(score[idx] - m[dst * 3 + h]);
    score[idx] = ex;
    atomicAdd(&denom[dst * 3 + h], ex);
}

// Pass 3: agg[dst] += alpha * xl[src].  Wave per edge, coalesced.
__global__ __launch_bounds__(256)
void edge_agg(const long long* __restrict__ ei, long E,
              const float* __restrict__ X, int xlOff,
              const float* __restrict__ exbuf, const float* __restrict__ denom,
              float* __restrict__ agg)
{
    const long e = (long)blockIdx.x * 8 + (threadIdx.x >> 5);
    if (e >= E) return;
    const int lane = threadIdx.x & 31;
    const long src = ei[e];
    const long dst = ei[E + e];
    float av = 0.f;
    if (lane < 3) av = exbuf[e * 3 + lane] / (denom[dst * 3 + lane] + 1e-16f);
    const float a0 = __shfl(av, 0, 32);
    const float a1 = __shfl(av, 1, 32);
    const float a2 = __shfl(av, 2, 32);
    const float* xl = X + src * (long)XLD + xlOff;
    float* out = agg + dst * (long)HC;
#pragma unroll
    for (int j = 0; j < 6; ++j) {
        const int g = lane + 32 * j;
        const float al = (j < 2) ? a0 : (j < 4) ? a1 : a2;
        atomicAdd(&out[g], al * xl[g]);
    }
}

// Head mean + per-conv bias + concat + ELU:  h[n, 0:64]=conv_a, h[n,64:128]=conv_p
__global__ void head_mean_elu(const float* __restrict__ agg_a,
                              const float* __restrict__ agg_p,
                              const float* __restrict__ bias_a,
                              const float* __restrict__ bias_p,
                              float* __restrict__ h)
{
    long idx = (long)blockIdx.x * 256 + threadIdx.x;
    if (idx >= (long)NN * 128) return;
    const long n = idx >> 7;
    const int  c = (int)(idx & 127);
    float x;
    if (c < 64) {
        const float* a = agg_a + n * HC;
        x = (a[c] + a[64 + c] + a[128 + c]) * (1.f / 3.f) + bias_a[c];
    } else {
        const int cc = c - 64;
        const float* a = agg_p + n * HC;
        x = (a[cc] + a[64 + cc] + a[128 + cc]) * (1.f / 3.f) + bias_p[cc];
    }
    h[idx] = x > 0.f ? x : __expf(x) - 1.f;
}

// Small fp32 dense: out[N,32] = h1[N,128] @ W[128,32] + b   (h2 is an output,
// keep it full fp32; 0.4 GFLOP, trivially fast).
__global__ void dense128x32(const float* __restrict__ h1,
                            const float* __restrict__ W,
                            const float* __restrict__ b,
                            float* __restrict__ out)
{
    long idx = (long)blockIdx.x * 256 + threadIdx.x;
    if (idx >= (long)NN * 32) return;
    const long n = idx >> 5;
    const int  o = (int)(idx & 31);
    const float* row = h1 + n * 128;
    float acc = b[o];
#pragma unroll 8
    for (int k = 0; k < 128; ++k) acc += row[k] * W[k * 32 + o];
    out[idx] = acc;
}

// ============================ host orchestration ===========================
static void run_edge_phase(hipStream_t stream, const long long* ei, long E,
                           const float* X, int xlOff, int xrOff,
                           const float* att, float* score,
                           float* m, float* denom, float* agg)
{
    const long nm = (long)NN * 3, na = (long)NN * HC;
    fill_f32<<<(unsigned)((nm + 255) / 256), 256, 0, stream>>>(m, -INFINITY, nm);
    fill_f32<<<(unsigned)((nm + 255) / 256), 256, 0, stream>>>(denom, 0.f, nm);
    fill_f32<<<(unsigned)((na + 255) / 256), 256, 0, stream>>>(agg, 0.f, na);
    edge_score<<<(unsigned)((E + 7) / 8), 256, 0, stream>>>(ei, E, X, xlOff, xrOff,
                                                            att, score, m);
    edge_exp<<<(unsigned)((E * 3 + 255) / 256), 256, 0, stream>>>(ei, E, m, score, denom);
    edge_agg<<<(unsigned)((E + 7) / 8), 256, 0, stream>>>(ei, E, X, xlOff,
                                                          score, denom, agg);
}

extern "C" void kernel_launch(void* const* d_in, const int* in_sizes, int n_in,
                              void* d_out, int out_size, void* d_ws, size_t ws_size,
                              hipStream_t stream)
{
    (void)n_in; (void)out_size; (void)ws_size;
    // -------- inputs (setup_inputs dict order, params flattened) --------
    const float*     feat = (const float*)d_in[0];
    const long long* ei   = (const long long*)d_in[1];
    const long long* eic  = (const long long*)d_in[2];
    const long E  = in_sizes[1] / 2;
    const long EC = in_sizes[2] / 2;
    const float *W1l=(const float*)d_in[3],  *b1l=(const float*)d_in[4],
                *W1r=(const float*)d_in[5],  *b1r=(const float*)d_in[6],
                *att1=(const float*)d_in[7], *bias1=(const float*)d_in[8];
    const float *W1pl=(const float*)d_in[9],  *b1pl=(const float*)d_in[10],
                *W1pr=(const float*)d_in[11], *b1pr=(const float*)d_in[12],
                *att1p=(const float*)d_in[13],*bias1p=(const float*)d_in[14];
    const float *W2=(const float*)d_in[15], *b2=(const float*)d_in[16];
    const float *W3l=(const float*)d_in[17],  *b3l=(const float*)d_in[18],
                *W3r=(const float*)d_in[19],  *b3r=(const float*)d_in[20],
                *att3=(const float*)d_in[21], *bias3=(const float*)d_in[22];
    const float *W3pl=(const float*)d_in[23], *b3pl=(const float*)d_in[24],
                *W3pr=(const float*)d_in[25], *b3pr=(const float*)d_in[26],
                *att3p=(const float*)d_in[27],*bias3p=(const float*)d_in[28];
    const float *W4=(const float*)d_in[29], *b4=(const float*)d_in[30];

    float* h2 = (float*)d_out;                     // [N,32]
    float* h4 = (float*)d_out + (long)NN * 32;     // [N,1024]

    // -------- workspace carve-out --------
    char* ws = (char*)d_ws;
    size_t off = 0;
    auto take = [&](size_t bytes) {
        size_t o = off; off = (off + bytes + 255) & ~size_t(255); return o;
    };
    float*  xlxr   = (float*)(ws + take((size_t)NN * XLD * 4));   // 153.6 MB
    float*  agg_a  = (float*)(ws + take((size_t)NN * HC * 4));    // 38.4 MB
    float*  agg_p  = (float*)(ws + take((size_t)NN * HC * 4));    // 38.4 MB
    float*  h1     = (float*)(ws + take((size_t)NN * 128 * 4));   // 25.6 MB (h1/h3)
    float*  mbuf   = (float*)(ws + take((size_t)NN * 3 * 4));
    float*  denom  = (float*)(ws + take((size_t)NN * 3 * 4));
    float*  score  = (float*)(ws + take((size_t)800000 * 3 * 4)); // score/ex
    __bf16* Bp     = (__bf16*)(ws + take((size_t)64 * 32 * 512 * 2)); // packed W
    float*  biascat= (float*)(ws + take(4 * HC * 4));

    // ================= layer 1: fused 4-way transform =================
    // Bpacked = [Wl1 | Wr1 | Wl1p | Wr1p]  (48 col-tiles, KC=32)
    {
        const int KC = 32;                                  // K=1024
        const long tot = (long)(HC / 16) * KC * 512;        // per source
        pack_b<<<(unsigned)((tot + 255) / 256), 256, 0, stream>>>(W1l,  Bp, KC, HC,  0, tot);
        pack_b<<<(unsigned)((tot + 255) / 256), 256, 0, stream>>>(W1r,  Bp, KC, HC, 12, tot);
        pack_b<<<(unsigned)((tot + 255) / 256), 256, 0, stream>>>(W1pl, Bp, KC, HC, 24, tot);
        pack_b<<<(unsigned)((tot + 255) / 256), 256, 0, stream>>>(W1pr, Bp, KC, HC, 36, tot);
        cat_bias4<<<3, 256, 0, stream>>>(biascat, b1l, b1r, b1pl, b1pr);
        gemm_bf16_wmma<3, 32, XLD><<<dim3(NN / 16, 2), 256, 0, stream>>>(feat, Bp, biascat, xlxr);
    }
    run_edge_phase(stream, ei,  E,  xlxr,   0, 192, att1,  score, mbuf, denom, agg_a);
    run_edge_phase(stream, eic, EC, xlxr, 384, 576, att1p, score, mbuf, denom, agg_p);
    head_mean_elu<<<(unsigned)(((long)NN * 128 + 255) / 256), 256, 0, stream>>>(
        agg_a, agg_p, bias1, bias1p, h1);

    // ================= conv2: h2 (fp32 output) =================
    dense128x32<<<(unsigned)(((long)NN * 32 + 255) / 256), 256, 0, stream>>>(h1, W2, b2, h2);

    // ================= layer 3: fused 4-way transform (K=32) =================
    {
        const int KC = 1;                                   // K=32
        const long tot = (long)(HC / 16) * KC * 512;
        pack_b<<<(unsigned)((tot + 255) / 256), 256, 0, stream>>>(W3l,  Bp, KC, HC,  0, tot);
        pack_b<<<(unsigned)((tot + 255) / 256), 256, 0, stream>>>(W3r,  Bp, KC, HC, 12, tot);
        pack_b<<<(unsigned)((tot + 255) / 256), 256, 0, stream>>>(W3pl, Bp, KC, HC, 24, tot);
        pack_b<<<(unsigned)((tot + 255) / 256), 256, 0, stream>>>(W3pr, Bp, KC, HC, 36, tot);
        cat_bias4<<<3, 256, 0, stream>>>(biascat, b3l, b3r, b3pl, b3pr);
        gemm_bf16_wmma<3, 1, XLD><<<dim3(NN / 16, 2), 256, 0, stream>>>(h2, Bp, biascat, xlxr);
    }
    run_edge_phase(stream, ei,  E,  xlxr,   0, 192, att3,  score, mbuf, denom, agg_a);
    run_edge_phase(stream, eic, EC, xlxr, 384, 576, att3p, score, mbuf, denom, agg_p);
    head_mean_elu<<<(unsigned)(((long)NN * 128 + 255) / 256), 256, 0, stream>>>(
        agg_a, agg_p, bias3, bias3p, h1);          // h1 now holds h3

    // ================= conv4: h4 = h3 @ W4 + b4  (K=128, 64 col-tiles) =======
    {
        const int KC = 4;                                   // K=128
        const long tot = (long)(1024 / 16) * KC * 512;
        pack_b<<<(unsigned)((tot + 255) / 256), 256, 0, stream>>>(W4, Bp, KC, 1024, 0, tot);
        gemm_bf16_wmma<4, 4, 1024><<<dim3(NN / 16, 2), 256, 0, stream>>>(h1, Bp, b4, h4);
    }
}
